// MultiHeadAttention_63067299774592
// MI455X (gfx1250) — matrix operand order repaired and checked by reading, not compiled, not run on hardware
//
#include <hip/hip_runtime.h>
#include <stdint.h>

#define EMBED   1024
#define SEQ     2048
#define BATCH   2
#define HEADS   16
#define DH      64
#define MROWS   (BATCH*SEQ)          // 4096
#define INV_TEMP 0.03125f            // 1/sqrt(1024)

typedef __attribute__((ext_vector_type(16))) __bf16 v16bf;
typedef __attribute__((ext_vector_type(8)))  float  v8f;

union FragB16 { v16bf v; uint4 q[2]; };
union PackBF  { __bf16 h[16]; uint4 q[2]; };

// Native convert: lets the backend use v_cvt_pk_bf16_f32 instead of bit-twiddling.
static __device__ __forceinline__ unsigned short f2bf(float f) {
  union { __bf16 h; unsigned short u; } c;
  c.h = (__bf16)f;
  return c.u;
}

// Convert 16 consecutive global fp32 -> 16 bf16, store 32B into LDS.
static __device__ __forceinline__ void cvt_store16(unsigned short* dst, const float* src) {
  float4 f0 = ((const float4*)src)[0];
  float4 f1 = ((const float4*)src)[1];
  float4 f2 = ((const float4*)src)[2];
  float4 f3 = ((const float4*)src)[3];
  PackBF p;
  p.h[0]=(__bf16)f0.x;  p.h[1]=(__bf16)f0.y;  p.h[2]=(__bf16)f0.z;  p.h[3]=(__bf16)f0.w;
  p.h[4]=(__bf16)f1.x;  p.h[5]=(__bf16)f1.y;  p.h[6]=(__bf16)f1.z;  p.h[7]=(__bf16)f1.w;
  p.h[8]=(__bf16)f2.x;  p.h[9]=(__bf16)f2.y;  p.h[10]=(__bf16)f2.z; p.h[11]=(__bf16)f2.w;
  p.h[12]=(__bf16)f3.x; p.h[13]=(__bf16)f3.y; p.h[14]=(__bf16)f3.z; p.h[15]=(__bf16)f3.w;
  *(uint4*)dst       = p.q[0];
  *(uint4*)(dst + 8) = p.q[1];
}

// CDNA5 async DMA: global -> LDS, 32B per thread (2 x b128), tracked by ASYNCcnt.
static __device__ __forceinline__ void async_copy32(unsigned short* lds_dst,
                                                    const unsigned short* gsrc) {
  unsigned lds = (unsigned)(uintptr_t)lds_dst;                 // LDS byte offset
  unsigned long long g = (unsigned long long)(uintptr_t)gsrc;  // 64-bit global addr
  asm volatile("global_load_async_to_lds_b128 %0, %1, off"
               :: "v"(lds), "v"(g) : "memory");
  asm volatile("global_load_async_to_lds_b128 %0, %1, off offset:16"
               :: "v"(lds), "v"(g) : "memory");
}

static __device__ __forceinline__ void wait_async() {
  asm volatile("s_wait_asynccnt 0x0" ::: "memory");
}

// ---------------------------------------------------------------------------
// Shared WMMA GEMM mainloop: C(128x128) = A(128xK) * W(128xK)^T block tile.
// A row-major MxK (fp32 converted in-flight, or bf16 via async-to-LDS DMA),
// W row-major NxK fp32 (torch Linear weight).
// 256 threads = 8 waves (4 along M x 2 along N); each wave: 2x4 WMMA tiles.
// LDS row stride 40 bf16 (80B): 16B fragment reads are bank-conflict free.
// ---------------------------------------------------------------------------
template<bool AF32>
static __device__ __forceinline__ void gemm_mainloop(
    const void* Ag_, const float* __restrict__ Wg, int m0, int n0,
    unsigned short* Xs, unsigned short* Ws, v8f acc[2][4])
{
  const int tid   = threadIdx.x;
  const int lane  = tid & 31, wave = tid >> 5;
  const int lid   = lane & 15, lhalf = lane >> 4;
  const int rowbase = (wave >> 1) * 32, colbase = (wave & 1) * 64;
  const int lrow = tid >> 1, lcol = (tid & 1) * 16;

  for (int k0 = 0; k0 < EMBED; k0 += 32) {
    __syncthreads();
    // --- stage A tile (128x32) as bf16 ---
    if (AF32) {
      const float* src = (const float*)Ag_ + (size_t)(m0 + lrow) * EMBED + k0 + lcol;
      cvt_store16(Xs + lrow * 40 + lcol, src);
      if (k0 + 32 < EMBED) __builtin_prefetch(src + 32, 0, 0);
    } else {
      const unsigned short* src =
          (const unsigned short*)Ag_ + (size_t)(m0 + lrow) * EMBED + k0 + lcol;
      async_copy32(Xs + lrow * 40 + lcol, src);   // bf16 already: DMA, no VGPR trip
    }
    // --- stage W tile (128x32) as bf16 ---
    {
      const float* src = Wg + (size_t)(n0 + lrow) * EMBED + k0 + lcol;
      cvt_store16(Ws + lrow * 40 + lcol, src);
      if (k0 + 32 < EMBED) __builtin_prefetch(src + 32, 0, 0);
    }
    if (!AF32) wait_async();
    __syncthreads();

    FragB16 a[2];
#pragma unroll
    for (int i = 0; i < 2; ++i) {
      const unsigned short* p = Xs + (rowbase + 16 * i + lid) * 40 + lhalf * 8;
      a[i].q[0] = *(const uint4*)p;
      a[i].q[1] = *(const uint4*)(p + 16);
    }
    FragB16 b[4];
#pragma unroll
    for (int j = 0; j < 4; ++j) {
      const unsigned short* p = Ws + (colbase + 16 * j + lid) * 40 + lhalf * 8;
      b[j].q[0] = *(const uint4*)p;
      b[j].q[1] = *(const uint4*)(p + 16);
    }
#pragma unroll
    for (int i = 0; i < 2; ++i)
#pragma unroll
      for (int j = 0; j < 4; ++j)
        acc[i][j] = __builtin_amdgcn_wmma_f32_16x16x32_bf16(
            false, a[i].v, false, b[j].v, (short)0, acc[i][j], false, false);
  }
}

// ---------------------------------------------------------------------------
// Kernel 1: fused QKV projections. grid.z selects q/k/v.
// q -> bf16 [B,H,N,Dh] pre-scaled by 1/temperature
// k -> bf16 [B,H,N,Dh]
// v -> bf16 [B,H,Dh,N]  (transposed so the P*V GEMM B-operand is K-contiguous)
// ---------------------------------------------------------------------------
__global__ __launch_bounds__(256) void qkv_gemm_kernel(
    const float* __restrict__ x,
    const float* __restrict__ Wq, const float* __restrict__ bq,
    const float* __restrict__ Wk, const float* __restrict__ bk,
    const float* __restrict__ Wv, const float* __restrict__ bv,
    unsigned short* __restrict__ qb, unsigned short* __restrict__ kbuf,
    unsigned short* __restrict__ vt)
{
  __shared__ __align__(16) unsigned short Xs[128 * 40];
  __shared__ __align__(16) unsigned short Ws[128 * 40];
  const int z = blockIdx.z;
  const float* W    = (z == 0) ? Wq : (z == 1) ? Wk : Wv;
  const float* bias = (z == 0) ? bq : (z == 1) ? bk : bv;
  const int m0 = blockIdx.x * 128, n0 = blockIdx.y * 128;

  v8f acc[2][4];
#pragma unroll
  for (int i = 0; i < 2; ++i)
#pragma unroll
    for (int j = 0; j < 4; ++j)
      acc[i][j] = v8f{0.f, 0.f, 0.f, 0.f, 0.f, 0.f, 0.f, 0.f};

  gemm_mainloop<true>(x, W, m0, n0, Xs, Ws, acc);

  const int lane = threadIdx.x & 31, wave = threadIdx.x >> 5;
  const int lid = lane & 15, lhalf = lane >> 4;
  const int rowbase = (wave >> 1) * 32, colbase = (wave & 1) * 64;
#pragma unroll
  for (int i = 0; i < 2; ++i)
#pragma unroll
    for (int j = 0; j < 4; ++j)
#pragma unroll
      for (int r = 0; r < 8; ++r) {
        int m = m0 + rowbase + 16 * i + lhalf * 8 + r;
        int n = n0 + colbase + 16 * j + lid;
        float val = acc[i][j][r] + bias[n];
        int bidx = m >> 11, row = m & (SEQ - 1);
        int hh = n >> 6, d = n & (DH - 1);
        size_t bh = (size_t)(bidx * HEADS + hh);
        if (z == 0)
          qb[(bh * SEQ + row) * DH + d] = f2bf(val * INV_TEMP);
        else if (z == 1)
          kbuf[(bh * SEQ + row) * DH + d] = f2bf(val);
        else
          vt[(bh * DH + d) * SEQ + row] = f2bf(val);
      }
}

// Async-copy a 64x64 bf16 tile (source row stride = src_stride elems) into LDS
// (stride 72). 256 threads x 32B = 8KB tile, all via ASYNCcnt DMA.
static __device__ __forceinline__ void load_tile_async(
    unsigned short* dst, const unsigned short* src, size_t src_stride, int tid)
{
  int row = tid >> 2, q4 = (tid & 3) * 16;
  async_copy32(dst + row * 72 + q4, src + (size_t)row * src_stride + q4);
}

// ---------------------------------------------------------------------------
// Kernel 2: flash-style attention. One block = 64 queries of one (b,h).
// 8 waves: wave covers 16 rows x 32 cols (2 WMMA tiles for S, 2 for O).
// ---------------------------------------------------------------------------
__global__ __launch_bounds__(256) void attn_kernel(
    const unsigned short* __restrict__ qb, const unsigned short* __restrict__ kbuf,
    const unsigned short* __restrict__ vt, unsigned short* __restrict__ ctx)
{
  __shared__ __align__(16) unsigned short Qs[64 * 72];
  __shared__ __align__(16) unsigned short Ks[64 * 72];
  __shared__ __align__(16) unsigned short Vs[64 * 72];
  __shared__ __align__(16) unsigned short Ps[64 * 72];
  __shared__ float Sf[64 * 65];
  __shared__ float alphaS[64];
  __shared__ float lS[64];

  const int tid = threadIdx.x;
  const int bh = blockIdx.y;
  const int b = bh >> 4, h = bh & (HEADS - 1);
  const int i0 = blockIdx.x * 64;

  const unsigned short* Q = qb   + ((size_t)bh * SEQ + i0) * DH;
  const unsigned short* K = kbuf + (size_t)bh * SEQ * DH;
  const unsigned short* V = vt   + (size_t)bh * DH * SEQ;

  const int lane = tid & 31, wave = tid >> 5;
  const int lid = lane & 15, lhalf = lane >> 4;
  const int rows16 = (wave >> 1) * 16;   // query-row tile of this wave
  const int cols32 = (wave & 1) * 32;    // key/d column pair of tiles

  load_tile_async(Qs, Q, DH, tid);       // covered by first wait_async below

  float m_i = -INFINITY, l_i = 0.f;      // per-row stats, owned by tid < 64
  v8f o0 = {0.f, 0.f, 0.f, 0.f, 0.f, 0.f, 0.f, 0.f};
  v8f o1 = {0.f, 0.f, 0.f, 0.f, 0.f, 0.f, 0.f, 0.f};

  for (int kb0 = 0; kb0 < SEQ; kb0 += 64) {
    __syncthreads();                      // previous iteration done with Ks/Vs/Ps
    load_tile_async(Ks, K + (size_t)kb0 * DH, DH, tid);
    load_tile_async(Vs, V + kb0, SEQ, tid);
    wait_async();
    __syncthreads();

    // ---- S = (Q/temp) @ K^T over Dh (2 k-chunks of 32) ----
    v8f s0 = {0.f, 0.f, 0.f, 0.f, 0.f, 0.f, 0.f, 0.f};
    v8f s1 = {0.f, 0.f, 0.f, 0.f, 0.f, 0.f, 0.f, 0.f};
#pragma unroll
    for (int kk = 0; kk < DH; kk += 32) {
      FragB16 a;
      const unsigned short* pa = Qs + (rows16 + lid) * 72 + kk + lhalf * 8;
      a.q[0] = *(const uint4*)pa;
      a.q[1] = *(const uint4*)(pa + 16);
      FragB16 b0, b1;
      const unsigned short* pb0 = Ks + (cols32 + lid) * 72 + kk + lhalf * 8;
      const unsigned short* pb1 = Ks + (cols32 + 16 + lid) * 72 + kk + lhalf * 8;
      b0.q[0] = *(const uint4*)pb0; b0.q[1] = *(const uint4*)(pb0 + 16);
      b1.q[0] = *(const uint4*)pb1; b1.q[1] = *(const uint4*)(pb1 + 16);
      s0 = __builtin_amdgcn_wmma_f32_16x16x32_bf16(false, a.v, false, b0.v, (short)0, s0, false, false);
      s1 = __builtin_amdgcn_wmma_f32_16x16x32_bf16(false, a.v, false, b1.v, (short)0, s1, false, false);
    }
#pragma unroll
    for (int r = 0; r < 8; ++r) {
      int rw = rows16 + lhalf * 8 + r;
      Sf[rw * 65 + cols32 + lid]      = s0[r];
      Sf[rw * 65 + cols32 + 16 + lid] = s1[r];
    }
    __syncthreads();

    // ---- online softmax: one thread per query row ----
    if (tid < 64) {
      const float* Srow = Sf + tid * 65;
      float mloc = Srow[0];
#pragma unroll 4
      for (int j = 1; j < 64; ++j) mloc = fmaxf(mloc, Srow[j]);
      float mnew = fmaxf(m_i, mloc);
      float al = __expf(m_i - mnew);
      float ps = 0.f;
#pragma unroll 4
      for (int j = 0; j < 64; ++j) {
        float p = __expf(Srow[j] - mnew);
        ps += p;
        Ps[tid * 72 + j] = f2bf(p);
      }
      l_i = l_i * al + ps;
      m_i = mnew;
      alphaS[tid] = al;
    }
    __syncthreads();

    // ---- O = O*alpha + P @ V  (V already transposed: rows = d) ----
#pragma unroll
    for (int r = 0; r < 8; ++r) {
      float al = alphaS[rows16 + lhalf * 8 + r];
      o0[r] *= al;
      o1[r] *= al;
    }
#pragma unroll
    for (int kk = 0; kk < 64; kk += 32) {
      FragB16 a;
      const unsigned short* pa = Ps + (rows16 + lid) * 72 + kk + lhalf * 8;
      a.q[0] = *(const uint4*)pa;
      a.q[1] = *(const uint4*)(pa + 16);
      FragB16 b0, b1;
      const unsigned short* pb0 = Vs + (cols32 + lid) * 72 + kk + lhalf * 8;
      const unsigned short* pb1 = Vs + (cols32 + 16 + lid) * 72 + kk + lhalf * 8;
      b0.q[0] = *(const uint4*)pb0; b0.q[1] = *(const uint4*)(pb0 + 16);
      b1.q[0] = *(const uint4*)pb1; b1.q[1] = *(const uint4*)(pb1 + 16);
      o0 = __builtin_amdgcn_wmma_f32_16x16x32_bf16(false, a.v, false, b0.v, (short)0, o0, false, false);
      o1 = __builtin_amdgcn_wmma_f32_16x16x32_bf16(false, a.v, false, b1.v, (short)0, o1, false, false);
    }
  }

  if (tid < 64) lS[tid] = l_i;
  __syncthreads();

#pragma unroll
  for (int r = 0; r < 8; ++r) {
    int row = rows16 + lhalf * 8 + r;
    float inv = 1.0f / lS[row];
    size_t gm = (size_t)(b * SEQ + i0 + row);
    int c0 = h * DH + cols32 + lid;
    ctx[gm * EMBED + c0]      = f2bf(o0[r] * inv);
    ctx[gm * EMBED + c0 + 16] = f2bf(o1[r] * inv);
  }
}

// ---------------------------------------------------------------------------
// Kernel 3: out = ctx @ Wp^T + bp + x  (pre-LayerNorm, fp32 to workspace)
// ---------------------------------------------------------------------------
__global__ __launch_bounds__(256) void proj_gemm_kernel(
    const unsigned short* __restrict__ ctx, const float* __restrict__ Wp,
    const float* __restrict__ bp, const float* __restrict__ x,
    float* __restrict__ preLN)
{
  __shared__ __align__(16) unsigned short Xs[128 * 40];
  __shared__ __align__(16) unsigned short Ws[128 * 40];
  const int m0 = blockIdx.x * 128, n0 = blockIdx.y * 128;

  v8f acc[2][4];
#pragma unroll
  for (int i = 0; i < 2; ++i)
#pragma unroll
    for (int j = 0; j < 4; ++j)
      acc[i][j] = v8f{0.f, 0.f, 0.f, 0.f, 0.f, 0.f, 0.f, 0.f};

  gemm_mainloop<false>(ctx, Wp, m0, n0, Xs, Ws, acc);

  const int lane = threadIdx.x & 31, wave = threadIdx.x >> 5;
  const int lid = lane & 15, lhalf = lane >> 4;
  const int rowbase = (wave >> 1) * 32, colbase = (wave & 1) * 64;
#pragma unroll
  for (int i = 0; i < 2; ++i)
#pragma unroll
    for (int j = 0; j < 4; ++j)
#pragma unroll
      for (int r = 0; r < 8; ++r) {
        int m = m0 + rowbase + 16 * i + lhalf * 8 + r;
        int n = n0 + colbase + 16 * j + lid;
        size_t off = (size_t)m * EMBED + n;
        preLN[off] = acc[i][j][r] + bp[n] + x[off];
      }
}

// ---------------------------------------------------------------------------
// Kernel 4: LayerNorm over the 1024-wide rows.
// ---------------------------------------------------------------------------
__global__ __launch_bounds__(256) void ln_kernel(
    const float* __restrict__ preLN, const float* __restrict__ gamma,
    const float* __restrict__ beta, float* __restrict__ out)
{
  __shared__ float rs[256];
  __shared__ float rq[256];
  const int row = blockIdx.x, tid = threadIdx.x;
  const float* src = preLN + (size_t)row * EMBED;
  float v[4], s = 0.f, q = 0.f;
#pragma unroll
  for (int e = 0; e < 4; ++e) {
    float xv = src[tid + e * 256];
    v[e] = xv;
    s += xv;
    q += xv * xv;
  }
  rs[tid] = s;
  rq[tid] = q;
  __syncthreads();
  for (int off = 128; off > 0; off >>= 1) {
    if (tid < off) {
      rs[tid] += rs[tid + off];
      rq[tid] += rq[tid + off];
    }
    __syncthreads();
  }
  float mu = rs[0] * (1.0f / EMBED);
  float var = rq[0] * (1.0f / EMBED) - mu * mu;
  float rsig = rsqrtf(var + 1e-6f);
  float* dst = out + (size_t)row * EMBED;
#pragma unroll
  for (int e = 0; e < 4; ++e) {
    int c = tid + e * 256;
    dst[c] = (v[e] - mu) * rsig * gamma[c] + beta[c];
  }
}

// ---------------------------------------------------------------------------
extern "C" void kernel_launch(void* const* d_in, const int* in_sizes, int n_in,
                              void* d_out, int out_size, void* d_ws, size_t ws_size,
                              hipStream_t stream) {
  (void)in_sizes; (void)n_in; (void)out_size; (void)ws_size;
  const float* x     = (const float*)d_in[0];
  const float* Wq    = (const float*)d_in[1];
  const float* bq    = (const float*)d_in[2];
  const float* Wk    = (const float*)d_in[3];
  const float* bk    = (const float*)d_in[4];
  const float* Wv    = (const float*)d_in[5];
  const float* bv    = (const float*)d_in[6];
  const float* Wp    = (const float*)d_in[7];
  const float* bp    = (const float*)d_in[8];
  const float* gamma = (const float*)d_in[9];
  const float* beta  = (const float*)d_in[10];

  char* ws = (char*)d_ws;
  const size_t MB = 1024ull * 1024ull;
  unsigned short* qb   = (unsigned short*)(ws + 0);       // 8MB  bf16 [B,H,N,Dh]
  unsigned short* kbuf = (unsigned short*)(ws + 8 * MB);  // 8MB  bf16 [B,H,N,Dh]
  unsigned short* vt   = (unsigned short*)(ws + 16 * MB); // 8MB  bf16 [B,H,Dh,N]
  unsigned short* ctx  = (unsigned short*)(ws + 24 * MB); // 8MB  bf16 [B*N,C]
  float*          preLN = (float*)(ws + 0);               // 16MB fp32, reuses q/k region

  qkv_gemm_kernel<<<dim3(MROWS / 128, EMBED / 128, 3), 256, 0, stream>>>(
      x, Wq, bq, Wk, bk, Wv, bv, qb, kbuf, vt);
  attn_kernel<<<dim3(SEQ / 64, BATCH * HEADS), 256, 0, stream>>>(qb, kbuf, vt, ctx);
  proj_gemm_kernel<<<dim3(MROWS / 128, EMBED / 128), 256, 0, stream>>>(
      ctx, Wp, bp, x, preLN);
  ln_kernel<<<dim3(MROWS), 256, 0, stream>>>(preLN, gamma, beta, (float*)d_out);
}
